// PointNet2EncoderModel_8710193676883
// MI455X (gfx1250) — matrix operand order repaired and checked by reading, not compiled, not run on hardware
//
#include <hip/hip_runtime.h>
#include <hip/hip_bf16.h>

// ---------------------------------------------------------------------------
// PointNet++ encoder for MI455X (gfx1250).
// GEMM layers use V_WMMA_F32_16X16X32_F16 (wave32). Each wave computes a
// 16x64 output tile: one A fragment is reused across 4 B fragments per
// K-step (4x less A traffic, 4 WMMAs per loop body).
// BN statistics (sum / sumsq per channel) are accumulated in the GEMM
// epilogue via LDS f32 atomics + one global atomic per (block, column).
// FPS runs one workgroup per cloud with xyz + distance arrays in LDS.
// ---------------------------------------------------------------------------

typedef __attribute__((ext_vector_type(16))) _Float16 v16h;
typedef __attribute__((ext_vector_type(8)))  _Float16 v8h;
typedef __attribute__((ext_vector_type(8)))  float    v8f;

#define BN_EPS 1e-5f

// ---------------------------------------------------------------------------
// Farthest point sampling: one block per batch. xyz (B,N,3) f32.
// Emits `far` BEFORE updating (matches jax.lax.scan emitting the carry).
// ---------------------------------------------------------------------------
__global__ void fps_kernel(const float* __restrict__ xyz, int* __restrict__ out_idx,
                           int N, int npoint) {
    __shared__ float sx[2048], sy[2048], sz[2048], sd[2048];
    __shared__ float rv[256];
    __shared__ int   ri[256];
    const int tid = threadIdx.x;
    const float* bxyz = xyz + (size_t)blockIdx.x * N * 3;
    for (int i = tid; i < N; i += 256) {
        sx[i] = bxyz[3 * i + 0];
        sy[i] = bxyz[3 * i + 1];
        sz[i] = bxyz[3 * i + 2];
        sd[i] = 1e10f;                    // reference init 1e10
    }
    __syncthreads();
    int far = 0;
    for (int it = 0; it < npoint; ++it) {
        if (tid == 0) out_idx[(size_t)blockIdx.x * npoint + it] = far;
        const float cx = sx[far], cy = sy[far], cz = sz[far];
        float bv = -1.0f; int bi = 0;
        for (int i = tid; i < N; i += 256) {
            float dx = sx[i] - cx, dy = sy[i] - cy, dz = sz[i] - cz;
            float d  = dx * dx + dy * dy + dz * dz;
            float nd = fminf(sd[i], d);
            sd[i] = nd;
            if (nd > bv) { bv = nd; bi = i; }   // ascending scan -> first-max tiebreak
        }
        rv[tid] = bv; ri[tid] = bi;
        __syncthreads();
        for (int off = 128; off > 0; off >>= 1) {
            if (tid < off) {
                if (rv[tid + off] > rv[tid] ||
                    (rv[tid + off] == rv[tid] && ri[tid + off] < ri[tid])) {
                    rv[tid] = rv[tid + off]; ri[tid] = ri[tid + off];
                }
            }
            __syncthreads();
        }
        far = ri[0];
        __syncthreads();   // everyone read ri[0] before it is rewritten
    }
}

// new_xyz[i,:] = xyz[b, idx[i], :]   (i over B*S)
__global__ void gather_centroids(const float* __restrict__ xyz, const int* __restrict__ idx,
                                 float* __restrict__ out, int N, int total, int S) {
    int i = blockIdx.x * blockDim.x + threadIdx.x;
    if (i >= total) return;
    int b = i / S;
    const float* src = xyz + ((size_t)b * N + idx[i]) * 3;
    out[3 * i + 0] = src[0];
    out[3 * i + 1] = src[1];
    out[3 * i + 2] = src[2];
}

// ---------------------------------------------------------------------------
// Ball query + group: one thread per (b,s) query. Scans points in ascending
// index order (== reference's sort + truncate), pads with first hit.
// Writes grouped features as f16 rows: [g_xyz - center (3), pts (Cp)] zero
// padded to Kp columns. Row index = (b*S+s)*K + j.
// ---------------------------------------------------------------------------
__global__ void ball_group(const float* __restrict__ xyz, const float* __restrict__ new_xyz,
                           const _Float16* __restrict__ pts, _Float16* __restrict__ Aout,
                           int B_, int N, int S, int K, int Cp, int Kp, float r2) {
    int i = blockIdx.x * blockDim.x + threadIdx.x;
    if (i >= B_ * S) return;
    int b = i / S;
    const float cx = new_xyz[3 * i + 0], cy = new_xyz[3 * i + 1], cz = new_xyz[3 * i + 2];
    const float* base = xyz + (size_t)b * N * 3;
    const size_t rowbase = (size_t)i * K;

    auto write_row = [&](int slot, int p) {
        _Float16* row = Aout + (rowbase + slot) * (size_t)Kp;
        row[0] = (_Float16)(base[3 * p + 0] - cx);
        row[1] = (_Float16)(base[3 * p + 1] - cy);
        row[2] = (_Float16)(base[3 * p + 2] - cz);
        if (pts) {
            const _Float16* ps = pts + ((size_t)b * N + p) * Cp;
            for (int c = 0; c < Cp; ++c) row[3 + c] = ps[c];
        }
        for (int c = 3 + Cp; c < Kp; ++c) row[c] = (_Float16)0.f;
    };

    int cnt = 0, first = 0;
    for (int p = 0; p < N && cnt < K; ++p) {
        float dx = base[3 * p + 0] - cx, dy = base[3 * p + 1] - cy, dz = base[3 * p + 2] - cz;
        if (dx * dx + dy * dy + dz * dz <= r2) {
            if (cnt == 0) first = p;
            write_row(cnt, p);
            ++cnt;
        }
    }
    for (; cnt < K; ++cnt) write_row(cnt, first);   // pad with first in-radius index
}

// group_all for SA3: row r = b*K + k -> [l2_xyz (3, NOT centered), l2_points (Cp)] pad to Kp
__global__ void group_all(const float* __restrict__ xyz, const _Float16* __restrict__ pts,
                          _Float16* __restrict__ Aout, int K, int Cp, int Kp, int total) {
    int r = blockIdx.x * blockDim.x + threadIdx.x;
    if (r >= total) return;
    _Float16* row = Aout + (size_t)r * Kp;
    row[0] = (_Float16)xyz[3 * r + 0];
    row[1] = (_Float16)xyz[3 * r + 1];
    row[2] = (_Float16)xyz[3 * r + 2];
    const _Float16* ps = pts + (size_t)r * Cp;
    for (int c = 0; c < Cp; ++c) row[3 + c] = ps[c];
    for (int c = 3 + Cp; c < Kp; ++c) row[c] = (_Float16)0.f;
}

// Weight convert f32 (Cout,Cin) -> f16 (Cout,Kp) zero padded
__global__ void wconv(const float* __restrict__ W, _Float16* __restrict__ Wh,
                      int Cin, int Kp, int total) {
    int i = blockIdx.x * blockDim.x + threadIdx.x;
    if (i >= total) return;
    int o = i / Kp, c = i % Kp;
    Wh[i] = (c < Cin) ? (_Float16)W[(size_t)o * Cin + c] : (_Float16)0.f;
}

// ---------------------------------------------------------------------------
// WMMA GEMM: Y[r,o] = sum_c A[r,c] * W[o,c]   (A: R x Kp f16, W: Cout x Kp f16)
// Block = 128 threads (4 waves); wave w -> rows m0=(by*4+w)*16, block cols
// n0 = bx*64; each wave computes 16x64 (4 accumulators, A reused 4x).
// Requires R % 64 == 0, Kp % 32 == 0, Cout % 64 == 0 (all true here).
// A-fragment layout: lane L (m = L&15, half = L>>4) holds
// K = half*16 + 0..15  -> two contiguous b128 loads.
// B-fragment uses the same contiguous K order as A (not AMD's table order):
// lane L (n = L&15, half = L>>4) holds K = half*16 + 0..15.
// ---------------------------------------------------------------------------
template <bool STATS, bool OUT16, bool OUT32, bool BIAS>
__global__ void gemm_wmma(const _Float16* __restrict__ A, const _Float16* __restrict__ W,
                          const float* __restrict__ bias,
                          _Float16* __restrict__ Y16, float* __restrict__ Y32,
                          float* __restrict__ gsum, float* __restrict__ gsq,
                          int Kp, int Cout) {
    __shared__ float cs[64], cq[64];
    const int lane = threadIdx.x & 31;
    const int wave = threadIdx.x >> 5;
    const int half = lane >> 4;
    const int l16  = lane & 15;
    const int m0 = (blockIdx.y * 4 + wave) * 16;
    const int n0 = blockIdx.x * 64;

    const _Float16* arow = A + (size_t)(m0 + l16) * Kp + half * 16;
    const _Float16* wrow0 = W + (size_t)(n0 +  0 + l16) * Kp + half * 16;
    const _Float16* wrow1 = W + (size_t)(n0 + 16 + l16) * Kp + half * 16;
    const _Float16* wrow2 = W + (size_t)(n0 + 32 + l16) * Kp + half * 16;
    const _Float16* wrow3 = W + (size_t)(n0 + 48 + l16) * Kp + half * 16;

    v8f acc0 = {}, acc1 = {}, acc2 = {}, acc3 = {};
    for (int k0 = 0; k0 < Kp; k0 += 32) {
        __builtin_prefetch(arow + k0 + 128, 0, 0);   // -> global_prefetch_b8
        v8h a0 = *(const v8h*)(arow + k0);
        v8h a1 = *(const v8h*)(arow + k0 + 8);
        v16h a = __builtin_shufflevector(a0, a1, 0,1,2,3,4,5,6,7,8,9,10,11,12,13,14,15);

        v8h b0 = *(const v8h*)(wrow0 + k0);
        v8h b1 = *(const v8h*)(wrow0 + k0 + 8);
        v16h b = __builtin_shufflevector(b0, b1, 0,1,2,3,4,5,6,7,8,9,10,11,12,13,14,15);
        acc0 = __builtin_amdgcn_wmma_f32_16x16x32_f16(false, a, false, b, (short)0, acc0, false, false);

        b0 = *(const v8h*)(wrow1 + k0);
        b1 = *(const v8h*)(wrow1 + k0 + 8);
        b = __builtin_shufflevector(b0, b1, 0,1,2,3,4,5,6,7,8,9,10,11,12,13,14,15);
        acc1 = __builtin_amdgcn_wmma_f32_16x16x32_f16(false, a, false, b, (short)0, acc1, false, false);

        b0 = *(const v8h*)(wrow2 + k0);
        b1 = *(const v8h*)(wrow2 + k0 + 8);
        b = __builtin_shufflevector(b0, b1, 0,1,2,3,4,5,6,7,8,9,10,11,12,13,14,15);
        acc2 = __builtin_amdgcn_wmma_f32_16x16x32_f16(false, a, false, b, (short)0, acc2, false, false);

        b0 = *(const v8h*)(wrow3 + k0);
        b1 = *(const v8h*)(wrow3 + k0 + 8);
        b = __builtin_shufflevector(b0, b1, 0,1,2,3,4,5,6,7,8,9,10,11,12,13,14,15);
        acc3 = __builtin_amdgcn_wmma_f32_16x16x32_f16(false, a, false, b, (short)0, acc3, false, false);
    }

    // D layout: VGPR i -> row m0 + i + half*8, col n0 + j*16 + l16
    v8f accs[4] = {acc0, acc1, acc2, acc3};
    float s[4], sq[4];
    for (int j = 0; j < 4; ++j) {
        const int col = n0 + j * 16 + l16;
        s[j] = 0.f; sq[j] = 0.f;
        for (int i = 0; i < 8; ++i) {
            const int row = m0 + i + half * 8;
            const float v = accs[j][i];
            if (OUT16) Y16[(size_t)row * Cout + col] = (_Float16)v;
            if (OUT32) {
                float o = v + (BIAS ? bias[col] : 0.f);
                Y32[(size_t)row * Cout + col] = o;
            }
            s[j] += v; sq[j] += v * v;
        }
    }
    if (STATS) {
        if (threadIdx.x < 64) { cs[threadIdx.x] = 0.f; cq[threadIdx.x] = 0.f; }
        __syncthreads();
        for (int j = 0; j < 4; ++j) {
            atomicAdd(&cs[j * 16 + l16], s[j]);      // ds_add_f32
            atomicAdd(&cq[j * 16 + l16], sq[j]);
        }
        __syncthreads();
        if (threadIdx.x < 64) {
            atomicAdd(&gsum[n0 + threadIdx.x], cs[threadIdx.x]);  // global_atomic_add_f32
            atomicAdd(&gsq[n0 + threadIdx.x],  cq[threadIdx.x]);
        }
    }
}

__global__ void bn_finalize(const float* __restrict__ gsum, const float* __restrict__ gsq,
                            float* __restrict__ mu, float* __restrict__ rs,
                            int C, float invR) {
    int c = blockIdx.x * blockDim.x + threadIdx.x;
    if (c >= C) return;
    float m = gsum[c] * invR;
    float v = gsq[c] * invR - m * m;
    mu[c] = m;
    rs[c] = rsqrtf(v + BN_EPS);
}

// normalized = relu(g*(x-mu)*rs + b), f16 -> f16
__global__ void bn_relu(const _Float16* __restrict__ X, _Float16* __restrict__ Y,
                        const float* __restrict__ mu, const float* __restrict__ rs,
                        const float* __restrict__ g, const float* __restrict__ b,
                        int C, size_t total) {
    size_t i = (size_t)blockIdx.x * blockDim.x + threadIdx.x;
    if (i >= total) return;
    int c = (int)(i % C);
    float x = (float)X[i];
    float y = g[c] * (x - mu[c]) * rs[c] + b[c];
    Y[i] = (_Float16)fmaxf(y, 0.f);
}

// out[s,c] = relu(max_k bn(X[(s*K+k), c]))   (relu is monotone -> relu after max)
__global__ void bn_relu_maxpool(const _Float16* __restrict__ X,
                                const float* __restrict__ mu, const float* __restrict__ rs,
                                const float* __restrict__ g, const float* __restrict__ b,
                                _Float16* __restrict__ O, int K, int C, int total) {
    int i = blockIdx.x * blockDim.x + threadIdx.x;
    if (i >= total) return;
    int s = i / C, c = i % C;
    const _Float16* base = X + (size_t)s * K * C + c;
    const float gm = g[c], bt = b[c], m0 = mu[c], r0 = rs[c];
    float best = -1e30f;
    for (int k = 0; k < K; ++k) {
        float x = (float)base[(size_t)k * C];
        best = fmaxf(best, gm * (x - m0) * r0 + bt);
    }
    O[i] = (_Float16)fmaxf(best, 0.f);
}

// Head BN (over axis 0, R samples) + exact GELU, f32 (R,C) -> f16 (R,C)
__global__ void head_bn_gelu(const float* __restrict__ H, const float* __restrict__ g,
                             const float* __restrict__ b, _Float16* __restrict__ Aout,
                             int R, int C) {
    int c = blockIdx.x * blockDim.x + threadIdx.x;
    if (c >= C) return;
    float s = 0.f, sq = 0.f;
    for (int r = 0; r < R; ++r) { float x = H[(size_t)r * C + c]; s += x; sq += x * x; }
    float m  = s / R;
    float rs = rsqrtf(sq / R - m * m + BN_EPS);
    for (int r = 0; r < R; ++r) {
        float y  = g[c] * (H[(size_t)r * C + c] - m) * rs + b[c];
        float ge = 0.5f * y * (1.f + erff(y * 0.70710678118654752f));
        Aout[(size_t)r * C + c] = (_Float16)ge;
    }
}

// ---------------------------------------------------------------------------
// Host orchestration
// ---------------------------------------------------------------------------
static void mlp_gemm(const float* W, int Cin, int Kp, int Cout, int R,
                     const _Float16* A, _Float16* Y16, _Float16* wbuf,
                     float* stats, float* mu, float* rs, hipStream_t stream) {
    int wtot = Cout * Kp;
    wconv<<<(wtot + 255) / 256, 256, 0, stream>>>(W, wbuf, Cin, Kp, wtot);
    hipMemsetAsync(stats, 0, 2048 * sizeof(float), stream);
    dim3 grid((unsigned)(Cout / 64), (unsigned)(R / 64));
    gemm_wmma<true, true, false, false><<<grid, 128, 0, stream>>>(
        A, wbuf, nullptr, Y16, nullptr, stats, stats + 1024, Kp, Cout);
    bn_finalize<<<(Cout + 255) / 256, 256, 0, stream>>>(
        stats, stats + 1024, mu, rs, Cout, 1.0f / (float)R);
}

extern "C" void kernel_launch(void* const* d_in, const int* in_sizes, int n_in,
                              void* d_out, int out_size, void* d_ws, size_t ws_size,
                              hipStream_t stream) {
    const int B = 64;                 // 4*16 clouds
    const float* points = (const float*)d_in[0];

    // SA params: 9 x (W, gamma, beta), then fc1_w, fc1_b, bn1_g, bn1_b, fc3_w, fc3_b
    const float* Wl[9]; const float* Gl[9]; const float* Bl[9];
    for (int l = 0; l < 9; ++l) {
        Wl[l] = (const float*)d_in[1 + 3 * l + 0];
        Gl[l] = (const float*)d_in[1 + 3 * l + 1];
        Bl[l] = (const float*)d_in[1 + 3 * l + 2];
    }
    const float* fc1_w = (const float*)d_in[28];
    const float* fc1_b = (const float*)d_in[29];
    const float* bn1_g = (const float*)d_in[30];
    const float* bn1_b = (const float*)d_in[31];
    const float* fc3_w = (const float*)d_in[32];
    const float* fc3_b = (const float*)d_in[33];

    // Workspace layout (~440 MB)
    size_t off = 0;
    char* wsb = (char*)d_ws;
    auto take = [&](size_t bytes) -> char* {
        char* p = wsb + off;
        off += (bytes + 255) & ~(size_t)255;
        return p;
    };
    _Float16* bufA   = (_Float16*)take((size_t)524288 * 160 * 2); // grouped / normalized acts
    _Float16* bufB   = (_Float16*)take((size_t)1048576 * 128 * 2); // pre-BN GEMM out
    int*      fps1   = (int*)take((size_t)B * 512 * 4);
    int*      fps2   = (int*)take((size_t)B * 128 * 4);
    float*    l1xyz  = (float*)take((size_t)B * 512 * 3 * 4);
    float*    l2xyz  = (float*)take((size_t)B * 128 * 3 * 4);
    _Float16* l1pts  = (_Float16*)take((size_t)B * 512 * 128 * 2);
    _Float16* l2pts  = (_Float16*)take((size_t)B * 128 * 256 * 2);
    _Float16* l3feat = (_Float16*)take((size_t)B * 1024 * 2);
    _Float16* wbuf   = (_Float16*)take((size_t)512 * 1024 * 2);
    float*    stats  = (float*)take(2048 * 4);
    float*    mu     = (float*)take(1024 * 4);
    float*    rs     = (float*)take(1024 * 4);
    float*    headH  = (float*)take((size_t)B * 512 * 4);
    _Float16* headA  = (_Float16*)take((size_t)B * 512 * 2);
    (void)ws_size; (void)in_sizes; (void)n_in; (void)out_size;

    // ---------------- SA1: N=2048 -> S=512, K=32, r=0.2, MLP 3->64->64->128
    fps_kernel<<<B, 256, 0, stream>>>(points, fps1, 2048, 512);
    gather_centroids<<<(B * 512 + 255) / 256, 256, 0, stream>>>(points, fps1, l1xyz, 2048, B * 512, 512);
    ball_group<<<(B * 512 + 255) / 256, 256, 0, stream>>>(points, l1xyz, nullptr, bufA,
                                                          B, 2048, 512, 32, 0, 32, 0.04f);
    {
        const int R = B * 512 * 32;   // 1,048,576
        const int Kp[3]   = {32, 64, 64};
        const int Cin[3]  = {3, 64, 64};
        const int Cout[3] = {64, 64, 128};
        for (int l = 0; l < 3; ++l) {
            mlp_gemm(Wl[l], Cin[l], Kp[l], Cout[l], R, bufA, bufB, wbuf, stats, mu, rs, stream);
            if (l < 2) {
                size_t tot = (size_t)R * Cout[l];
                bn_relu<<<(unsigned)((tot + 255) / 256), 256, 0, stream>>>(
                    bufB, bufA, mu, rs, Gl[l], Bl[l], Cout[l], tot);
            } else {
                int tot = B * 512 * 128;
                bn_relu_maxpool<<<(tot + 255) / 256, 256, 0, stream>>>(
                    bufB, mu, rs, Gl[l], Bl[l], l1pts, 32, 128, tot);
            }
        }
    }

    // ---------------- SA2: N=512 -> S=128, K=64, r=0.4, MLP 131->128->128->256
    fps_kernel<<<B, 256, 0, stream>>>(l1xyz, fps2, 512, 128);
    gather_centroids<<<(B * 128 + 255) / 256, 256, 0, stream>>>(l1xyz, fps2, l2xyz, 512, B * 128, 128);
    ball_group<<<(B * 128 + 255) / 256, 256, 0, stream>>>(l1xyz, l2xyz, l1pts, bufA,
                                                          B, 512, 128, 64, 128, 160, 0.16f);
    {
        const int R = B * 128 * 64;   // 524,288
        const int Kp[3]   = {160, 128, 128};
        const int Cin[3]  = {131, 128, 128};
        const int Cout[3] = {128, 128, 256};
        for (int l = 0; l < 3; ++l) {
            mlp_gemm(Wl[3 + l], Cin[l], Kp[l], Cout[l], R, bufA, bufB, wbuf, stats, mu, rs, stream);
            if (l < 2) {
                size_t tot = (size_t)R * Cout[l];
                bn_relu<<<(unsigned)((tot + 255) / 256), 256, 0, stream>>>(
                    bufB, bufA, mu, rs, Gl[3 + l], Bl[3 + l], Cout[l], tot);
            } else {
                int tot = B * 128 * 256;
                bn_relu_maxpool<<<(tot + 255) / 256, 256, 0, stream>>>(
                    bufB, mu, rs, Gl[3 + l], Bl[3 + l], l2pts, 64, 256, tot);
            }
        }
    }

    // ---------------- SA3: group_all over 128 pts, MLP 259->256->512->1024
    group_all<<<(B * 128 + 255) / 256, 256, 0, stream>>>(l2xyz, l2pts, bufA, 128, 256, 288, B * 128);
    {
        const int R = B * 128;        // 8192
        const int Kp[3]   = {288, 256, 512};
        const int Cin[3]  = {259, 256, 512};
        const int Cout[3] = {256, 512, 1024};
        for (int l = 0; l < 3; ++l) {
            mlp_gemm(Wl[6 + l], Cin[l], Kp[l], Cout[l], R, bufA, bufB, wbuf, stats, mu, rs, stream);
            if (l < 2) {
                size_t tot = (size_t)R * Cout[l];
                bn_relu<<<(unsigned)((tot + 255) / 256), 256, 0, stream>>>(
                    bufB, bufA, mu, rs, Gl[6 + l], Bl[6 + l], Cout[l], tot);
            } else {
                int tot = B * 1024;
                bn_relu_maxpool<<<(tot + 255) / 256, 256, 0, stream>>>(
                    bufB, mu, rs, Gl[6 + l], Bl[6 + l], l3feat, 128, 1024, tot);
            }
        }
    }

    // ---------------- Head: fc1 (1024->512) + BN + GELU + fc3 (512->128)
    wconv<<<(512 * 1024 + 255) / 256, 256, 0, stream>>>(fc1_w, wbuf, 1024, 1024, 512 * 1024);
    gemm_wmma<false, false, true, true><<<dim3(512 / 64, 64 / 64), 128, 0, stream>>>(
        l3feat, wbuf, fc1_b, nullptr, headH, nullptr, nullptr, 1024, 512);
    head_bn_gelu<<<2, 256, 0, stream>>>(headH, bn1_g, bn1_b, headA, 64, 512);
    wconv<<<(128 * 512 + 255) / 256, 256, 0, stream>>>(fc3_w, wbuf, 512, 512, 128 * 512);
    gemm_wmma<false, false, true, true><<<dim3(128 / 64, 64 / 64), 128, 0, stream>>>(
        headA, wbuf, fc3_b, nullptr, (float*)d_out, nullptr, nullptr, 512, 128);
}
